// RNN_11278584119807
// MI455X (gfx1250) — matrix operand order repaired and checked by reading, not compiled, not run on hardware
//
#include <hip/hip_runtime.h>

// Problem constants (from reference)
#define TSEQ  2048
#define BATCH 512
#define HID   10
#define EMB   10
#define LAB   17
#define NTILE 32          // BATCH / 16 batch-tiles, one wave each
// Scratch layout ("C-layout order"): xp[t][g][lane][r] floats, flat index
//   ((t*NTILE + g)*32 + lane)*8 + r
// where lane = hi*16 + n (n = hidden column 0..15, hi = lane>>4) and the
// batch row within the tile is m = 8*hi + r.  This matches the WMMA 16x16
// f32 C/D VGPR layout exactly, so kernel B's per-step add is two b128 loads.
// Required ws bytes: TSEQ*NTILE*32*8*4 = 64 MiB.

typedef _Float16 v16h __attribute__((ext_vector_type(16)));
typedef _Float16 v8h  __attribute__((ext_vector_type(8)));
typedef float    v8f  __attribute__((ext_vector_type(8)));

__device__ __forceinline__ float fast_tanh(float x) {
    // tanh(x) = 1 - 2/(exp(2x)+1), via v_exp_f32 + v_rcp_f32 (no IEEE div
    // expansion, no libm).  Clamp compiles to v_min_num/v_max_num.
    x = __builtin_fmaxf(-10.f, __builtin_fminf(10.f, x));
    float e = __builtin_amdgcn_exp2f(x * 2.8853900817779268f); // 2*log2(e)
    float r = __builtin_amdgcn_rcpf(e + 1.f);
    return __builtin_fmaf(-2.f, r, 1.f);
}

// ---------------------------------------------------------------------------
// Kernel A: embedding gather + input projection, written in C-layout order.
// One thread per (t, g, lane); computes the 8 r-values (8 batch rows, one
// hidden column n) -> 32 contiguous bytes, perfectly coalesced across lanes.
// ---------------------------------------------------------------------------
__global__ void embed_project_kernel(const int*   __restrict__ x,
                                     const float* __restrict__ emb,
                                     const float* __restrict__ W_ih,
                                     const float* __restrict__ b_ih,
                                     const float* __restrict__ b_hh,
                                     float*       __restrict__ xp) {
    int idx = blockIdx.x * blockDim.x + threadIdx.x;   // TSEQ*32*32 threads
    int ln  = idx & 31;
    int g   = (idx >> 5) & 31;
    int t   = idx >> 10;
    if (t >= TSEQ) return;
    int n   = ln & 15;
    int hi  = ln >> 4;

    float out[8];
    if (n < HID) {
        float wrow[EMB];
#pragma unroll
        for (int e = 0; e < EMB; ++e) wrow[e] = W_ih[n * EMB + e];
        float bias = b_ih[n] + b_hh[n];
#pragma unroll
        for (int r = 0; r < 8; ++r) {
            int b = g * 16 + 8 * hi + r;
            int tok = x[(size_t)b * TSEQ + t];
            const float* er = emb + (size_t)tok * EMB;
            float acc = bias;
#pragma unroll
            for (int e = 0; e < EMB; ++e) acc = fmaf(er[e], wrow[e], acc);
            out[r] = acc;
        }
    } else {
#pragma unroll
        for (int r = 0; r < 8; ++r) out[r] = 0.f;      // K/N padding stays zero
    }

    float4* dst = (float4*)(xp + (size_t)idx * 8);
    dst[0] = make_float4(out[0], out[1], out[2], out[3]);
    dst[1] = make_float4(out[4], out[5], out[6], out[7]);
}

// ---------------------------------------------------------------------------
// Kernel B: the sequential scan.  One wave (32 lanes) per 16-batch tile.
// Per step: one v_wmma_f32_16x16x32_f16, add prefetched xp slice, tanh,
// store hs in place over xp, rebuild the f16 A-operand via an LDS transpose.
// ---------------------------------------------------------------------------
__global__ void __launch_bounds__(32)
rnn_scan_kernel(const float* __restrict__ W_hh,
                float*       __restrict__ xp /* in: xp, out (in place): hs */) {
    int g  = blockIdx.x;        // 0..31 batch tile
    int ln = threadIdx.x;       // 0..31 lane
    int n  = ln & 15;
    int hi = ln >> 4;

    // B operand: W_hh^T as 16-bit 32x16 (K x N).  VGPR r holds packed
    // K = {2r, 2r+1} + 8*hi (+8 if r>=4) for column N = n.  K>=10 or N>=10 -> 0.
    v16h bw;
#pragma unroll
    for (int r = 0; r < 8; ++r) {
        int k0 = 2 * r + 8 * hi + (r >= 4 ? 8 : 0);
        int k1 = k0 + 1;
        bw[2 * r]     = (n < HID && k0 < HID) ? (_Float16)W_hh[n * HID + k0]
                                              : (_Float16)0.f;
        bw[2 * r + 1] = (n < HID && k1 < HID) ? (_Float16)W_hh[n * HID + k1]
                                              : (_Float16)0.f;
    }

    // Per-wave LDS staging buffer: 16x16 f16 row-major (512 B).
    __shared__ __align__(16) _Float16 hbuf[256];

    // A operand: h in 16-bit 16x32 layout.  h0 = 0.
    v16h a;
#pragma unroll
    for (int i = 0; i < 16; ++i) a[i] = (_Float16)0.f;

    const size_t tstride = (size_t)NTILE * 32 * 8;             // floats per t
    float* cur0 = xp + ((size_t)g * 32 + ln) * 8;              // t = 0 slice

    float4 xa = *(const float4*)(cur0);
    float4 xb = *(const float4*)(cur0 + 4);

    for (int t = 0; t < TSEQ; ++t) {
        float* cur = cur0 + (size_t)t * tstride;

        // Double-buffer next step's xp; prefetch two steps ahead into L2/L0.
        float4 na = make_float4(0.f, 0.f, 0.f, 0.f);
        float4 nb = na;
        if (t + 1 < TSEQ) {
            const float* nxt = cur + tstride;
            na = *(const float4*)(nxt);
            nb = *(const float4*)(nxt + 4);
        }
        if (t + 2 < TSEQ) {
            __builtin_prefetch(cur + 2 * tstride, 0, 1);       // global_prefetch_b8
        }

        // D = h * W_hh^T  (16x16 f32, C layout)
        v8f c = {0.f, 0.f, 0.f, 0.f, 0.f, 0.f, 0.f, 0.f};
        c = __builtin_amdgcn_wmma_f32_16x16x32_f16(
                /*neg_a=*/false, a, /*neg_b=*/false, bw,
                /*c_mod=*/(short)0, c, /*reuse_a=*/false, /*reuse_b=*/false);

        float v[8];
        v[0] = fast_tanh(c[0] + xa.x);
        v[1] = fast_tanh(c[1] + xa.y);
        v[2] = fast_tanh(c[2] + xa.z);
        v[3] = fast_tanh(c[3] + xa.w);
        v[4] = fast_tanh(c[4] + xb.x);
        v[5] = fast_tanh(c[5] + xb.y);
        v[6] = fast_tanh(c[6] + xb.z);
        v[7] = fast_tanh(c[7] + xb.w);

        // Store hs in place over xp[t] (same C-layout addresses).
        *(float4*)(cur)     = make_float4(v[0], v[1], v[2], v[3]);
        *(float4*)(cur + 4) = make_float4(v[4], v[5], v[6], v[7]);

        // Repack C layout -> A layout through LDS.  Lane (hi,n) owns rows
        // m = 8*hi + r of column n; columns n>=10 hold tanh(0)=0, preserving
        // the K-padding invariant for the next WMMA.
#pragma unroll
        for (int r = 0; r < 8; ++r) {
            int m = 8 * hi + r;
            hbuf[m * 16 + n] = (_Float16)v[r];
        }
        asm volatile("s_wait_dscnt 0" ::: "memory");   // DS stores visible

        // A row M = ln&15 (= n); lane half selects K range 0..7 / 8..15.
        v8h lo = *(const v8h*)&hbuf[n * 16 + hi * 8];  // ds_load_b128
#pragma unroll
        for (int i = 0; i < 8; ++i)  a[i] = lo[i];
#pragma unroll
        for (int i = 8; i < 16; ++i) a[i] = (_Float16)0.f;   // K = 16..31
        asm volatile("" ::: "memory");

        xa = na;
        xb = nb;
    }
}

// ---------------------------------------------------------------------------
// Kernel C: logits = hs @ fc_W^T + fc_b in f32.  t-fastest thread mapping so
// the 17-float output rows are the coalesced stream (dominant traffic).
// ---------------------------------------------------------------------------
__global__ void logits_kernel(const float* __restrict__ hs,
                              const float* __restrict__ fc_W,
                              const float* __restrict__ fc_b,
                              float*       __restrict__ out) {
    int idx = blockIdx.x * blockDim.x + threadIdx.x;   // BATCH*TSEQ threads
    if (idx >= BATCH * TSEQ) return;
    int t = idx & (TSEQ - 1);
    int b = idx >> 11;                                 // TSEQ = 2048
    int g = b >> 4, m = b & 15, hi = m >> 3, r = m & 7;

    const float* basep = hs + ((size_t)t * NTILE + g) * 256 + (size_t)r;
    float h[HID];
#pragma unroll
    for (int nn = 0; nn < HID; ++nn)
        h[nn] = basep[(size_t)(hi * 16 + nn) * 8];

    float* o = out + (size_t)idx * LAB;
#pragma unroll
    for (int l = 0; l < LAB; ++l) {
        float acc = fc_b[l];
#pragma unroll
        for (int nn = 0; nn < HID; ++nn)
            acc = fmaf(h[nn], fc_W[l * HID + nn], acc);
        o[l] = acc;
    }
}

// ---------------------------------------------------------------------------
extern "C" void kernel_launch(void* const* d_in, const int* in_sizes, int n_in,
                              void* d_out, int out_size, void* d_ws, size_t ws_size,
                              hipStream_t stream) {
    const int*   x    = (const int*)  d_in[0];
    const float* emb  = (const float*)d_in[1];
    const float* W_ih = (const float*)d_in[2];
    const float* W_hh = (const float*)d_in[3];
    const float* b_ih = (const float*)d_in[4];
    const float* b_hh = (const float*)d_in[5];
    const float* fc_W = (const float*)d_in[6];
    const float* fc_b = (const float*)d_in[7];
    float* out = (float*)d_out;
    float* xp  = (float*)d_ws;   // needs TSEQ*NTILE*32*8*4 = 64 MiB scratch

    (void)in_sizes; (void)n_in; (void)out_size; (void)ws_size;

    // A: embedding + input projection into C-layout scratch
    {
        int total = TSEQ * 32 * 32;                    // 2,097,152
        embed_project_kernel<<<total / 256, 256, 0, stream>>>(
            x, emb, W_ih, b_ih, b_hh, xp);
    }
    // B: sequential scan, one wave per 16-batch tile, WMMA on critical path
    rnn_scan_kernel<<<NTILE, 32, 0, stream>>>(W_hh, xp);
    // C: output projection (f32), coalesced logits stores
    {
        int total = BATCH * TSEQ;                      // 1,048,576
        logits_kernel<<<total / 256, 256, 0, stream>>>(xp, fc_W, fc_b, out);
    }
}
